// ModulatedDeformConv_45646912422538
// MI455X (gfx1250) — matrix operand order repaired and checked by reading, not compiled, not run on hardware
//
#include <hip/hip_runtime.h>
#include <hip/hip_bf16.h>

// ---------------------------------------------------------------------------
// Modulated deformable conv (DCNv2) forward for gfx1250 (MI455X, wave32).
// Phase 1: bilinear gather -> fp16 im2col tile in LDS.
// Phase 2: v_wmma_f32_16x16x32_f16 GEMM over K = CIN*9 = 576.
// ---------------------------------------------------------------------------

typedef __attribute__((ext_vector_type(16))) _Float16 v16h;
typedef __attribute__((ext_vector_type(8)))  _Float16 v8h;
typedef __attribute__((ext_vector_type(8)))  float    v8f;

#define BN    8
#define CIN   64
#define HH    128
#define WW    128
#define COUT  64
#define TT    9            // 3x3 taps
#define HO    128
#define WO    128
#define HW    (HH * WW)
#define KDIM  (CIN * TT)   // 576
#define WP    72           // weight LDS row pitch (halves), padded vs bank conflicts
#define AP    592          // A-matrix LDS row pitch (halves), padded
#define MTILE 32           // output pixels per tile (consecutive wo, same b/ho)
#define NTILES (BN * HO * WO / MTILE)  // 4096
#define TPB   256          // 8 waves

// Dynamic LDS layout (bytes):
//   pw    : 288*4 floats  = 4608
//   pi    : 288*4 ints    = 4608
//   sbias : 64 floats     =  256
//   Wlds  : 576*72 halves = 82944
//   Alds  : 32*592 halves = 37888
#define OFF_PW   0
#define OFF_PI   4608
#define OFF_BIAS 9216
#define OFF_W    9472
#define OFF_A    (9472 + 82944)
#define SMEM_BYTES (OFF_A + 37888)   // 130304

__global__ __launch_bounds__(TPB) void
dcn2_wmma_kernel(const float* __restrict__ x,
                 const float* __restrict__ off,
                 const float* __restrict__ msk,
                 const float* __restrict__ wgt,
                 const float* __restrict__ bias,
                 float* __restrict__ out)
{
    extern __shared__ char smem[];
    float*    pw    = (float*)   (smem + OFF_PW);
    int*      pi    = (int*)     (smem + OFF_PI);
    float*    sbias = (float*)   (smem + OFF_BIAS);
    _Float16* Wl    = (_Float16*)(smem + OFF_W);
    _Float16* Al    = (_Float16*)(smem + OFF_A);

    const int tid  = threadIdx.x;
    const int lane = tid & 31;
    const int wv   = tid >> 5;   // wave id 0..7
    const int mt   = wv >> 2;    // M-tile 0..1 (16 pixels each)
    const int nt   = wv & 3;     // N-tile 0..3 (16 couts each)

    // ---- One-time: stage reordered fp16 weights into LDS -------------------
    // Global weight layout [o][c][t] (t = ky*3+kx); LDS layout Wl[k=t*64+c][o].
    for (int j = tid; j < COUT * KDIM; j += TPB) {        // coalesced reads
        const int o = j / KDIM;
        const int r = j - o * KDIM;
        const int c = r / TT;
        const int t = r - c * TT;
        Wl[(t * CIN + c) * WP + o] = (_Float16)wgt[j];
    }
    if (tid < COUT) sbias[tid] = bias[tid];
    __syncthreads();

    // ---- Persistent loop over pixel tiles ----------------------------------
    for (int tile = blockIdx.x; tile < NTILES; tile += gridDim.x) {
        const int gp0 = tile * MTILE;
        const int b   = gp0 / (HO * WO);
        const int rp  = gp0 - b * (HO * WO);
        const int ho  = rp / WO;
        const int wo0 = rp - ho * WO;    // 32 consecutive wo, same row

        // -- Stage 1: bilinear params for 32 pixels x 9 taps ----------------
        for (int j = tid; j < MTILE * TT; j += TPB) {
            const int p  = j / TT;
            const int t  = j - p * TT;
            const int wo = wo0 + p;
            const int ob = ((b * (2 * TT) + 2 * t) * HO + ho) * WO + wo;
            const float dy = off[ob];
            const float dx = off[ob + HO * WO];
            const float m  = msk[((b * TT + t) * HO + ho) * WO + wo];
            const float py = dy + (float)(ho - 1 + t / 3);
            const float px = dx + (float)(wo - 1 + t % 3);
            const float y0 = floorf(py);
            const float x0 = floorf(px);
            const float fy = py - y0;
            const float fx = px - x0;
            const int iy0 = (int)y0;
            const int ix0 = (int)x0;
            const int   ys[2] = { iy0, iy0 + 1 };
            const int   xs[2] = { ix0, ix0 + 1 };
            const float wy[2] = { 1.0f - fy, fy };
            const float wx[2] = { 1.0f - fx, fx };
#pragma unroll
            for (int q = 0; q < 4; ++q) {
                const int yy = ys[q >> 1];
                const int xx = xs[q & 1];
                const bool v = (yy >= 0) && (yy < HH) && (xx >= 0) && (xx < WW);
                const int yc = min(max(yy, 0), HH - 1);
                const int xc = min(max(xx, 0), WW - 1);
                pw[4 * j + q] = v ? (wy[q >> 1] * wx[q & 1] * m) : 0.0f;
                pi[4 * j + q] = yc * WW + xc;
            }
        }
        __syncthreads();

        // -- Stage 2: gather channels, build fp16 A[32][576] in LDS ---------
        {
            const float* xb = x + (size_t)b * CIN * HW;
            for (int j = tid; j < MTILE * TT; j += TPB) {
                const int p = j / TT;
                const int t = j - p * TT;
                const float w0 = pw[4 * j + 0], w1 = pw[4 * j + 1];
                const float w2 = pw[4 * j + 2], w3 = pw[4 * j + 3];
                const int   i0 = pi[4 * j + 0], i1 = pi[4 * j + 1];
                const int   i2 = pi[4 * j + 2], i3 = pi[4 * j + 3];
                unsigned* arow = (unsigned*)(Al + p * AP + t * CIN);
#pragma unroll 4
                for (int c = 0; c < CIN; c += 2) {
                    const float* xc0 = xb + (size_t)c * HW;
                    const float* xc1 = xc0 + HW;
                    const float a0 = w0 * xc0[i0] + w1 * xc0[i1] + w2 * xc0[i2] + w3 * xc0[i3];
                    const float a1 = w0 * xc1[i0] + w1 * xc1[i1] + w2 * xc1[i2] + w3 * xc1[i3];
                    union { _Float16 h[2]; unsigned u; } pk;
                    pk.h[0] = (_Float16)a0;
                    pk.h[1] = (_Float16)a1;
                    arow[c >> 1] = pk.u;
                }
            }
        }
        __syncthreads();

        // -- Stage 3: WMMA — each wave does one 16x16 D-tile, K loop 576 ----
        v8f acc = {};
        const int arow0 = mt * 16 + (lane & 15);           // A row = M (lane%16)
        const int acol  = (lane < 16) ? 0 : 8;             // ISA A-layout K split
        const _Float16* Arow = Al + arow0 * AP;
#pragma unroll
        for (int kb = 0; kb < KDIM; kb += 32) {
            // A: lane holds halves K = kb + {acol..acol+7, acol+16..acol+23}
            v8h alo = *(const v8h*)(Arow + kb + acol);
            v8h ahi = *(const v8h*)(Arow + kb + acol + 16);
            v16h av = __builtin_shufflevector(alo, ahi,
                        0,1,2,3,4,5,6,7,8,9,10,11,12,13,14,15);
            // B: lane = N col (lane%16), halves K = kb + {acol..acol+7, acol+16..acol+23}
            const _Float16* brow = Wl + (kb + acol) * WP + nt * 16 + (lane & 15);
            v16h bv;
#pragma unroll
            for (int s = 0; s < 16; ++s)
                bv[s] = brow[(s < 8 ? s : s + 8) * WP];
            acc = __builtin_amdgcn_wmma_f32_16x16x32_f16(
                      false, av, false, bv, (short)0, acc, false, false);
        }

        // -- Store: lane holds M = v + (lane>=16?8:0), N = lane%16 ----------
        {
            const int o    = nt * 16 + (lane & 15);
            const int wo_s = wo0 + mt * 16 + ((lane < 16) ? 0 : 8);
            const float bo = sbias[o];
            float* op = out + ((size_t)(b * COUT + o) * HO + ho) * WO + wo_s;
#pragma unroll
            for (int v = 0; v < 8; ++v)
                op[v] = acc[v] + bo;   // 8 consecutive wo -> 32B contiguous
        }
        // No end-of-tile barrier needed: the stage1->stage2 barrier of the
        // next iteration orders this wave's A/W reads before the next writes.
    }
}

extern "C" void kernel_launch(void* const* d_in, const int* in_sizes, int n_in,
                              void* d_out, int out_size, void* d_ws, size_t ws_size,
                              hipStream_t stream)
{
    const float* x    = (const float*)d_in[0];
    const float* off  = (const float*)d_in[1];
    const float* msk  = (const float*)d_in[2];
    const float* wgt  = (const float*)d_in[3];
    const float* bias = (const float*)d_in[4];
    float* out = (float*)d_out;

    const int grid = 1024;   // persistent: each block processes 4 tiles
    hipLaunchKernelGGL(dcn2_wmma_kernel, dim3(grid), dim3(TPB), SMEM_BYTES, stream,
                       x, off, msk, wgt, bias, out);
}
